// GATNet_78529182040426
// MI455X (gfx1250) — matrix operand order repaired and checked by reading, not compiled, not run on hardware
//
#include <hip/hip_runtime.h>
#include <hip/hip_bf16.h>
#include <math.h>

// ---------------- problem constants (from reference) ----------------
#define NN      20000          // nodes
#define E0      320000         // original edges
#define E_TOT   340000         // + self loops
#define NG      512            // graphs
#define H1N     10             // heads layer1
#define C1N     78             // per-head channels layer1
#define D1N     780            // H1N*C1N
#define D1PAD   832            // N-pad (x64) for GEMM1 output (780 -> 832, 13 groups of 4 tiles)
#define K1PAD   96             // K-pad (x32) for GEMM1 (78 -> 96)
#define C2N     100            // layer2 channels (heads=1)
#define K2PAD   800            // K-pad for GEMM2 (780 -> 800)
#define N2PAD   128            // N-pad (x64) for GEMM2 (100 -> 128, 2 groups)
#define NTW     4              // N-tiles per wave in the GEMM (A-fragment reuse)
#define NEG_SLOPE 0.2f

typedef __attribute__((ext_vector_type(16))) __bf16 v16bf;
typedef __attribute__((ext_vector_type(8)))  float  v8f;

// ---------------- helpers ----------------
__device__ __forceinline__ unsigned short f2bf(float f) {
  unsigned int u = __float_as_uint(f);
  u += 0x7fffu + ((u >> 16) & 1u);           // round-to-nearest-even
  return (unsigned short)(u >> 16);
}

__device__ __forceinline__ void atomicMaxF(float* addr, float val) {
  float old = *addr;
  while (val > old) {
    int o = __float_as_int(old);
    int prev = atomicCAS((int*)addr, o, __float_as_int(val));
    if (prev == o) break;
    old = __int_as_float(prev);
  }
}

__device__ __forceinline__ void edge_sd(const int* __restrict__ ei, int e, int& s, int& d) {
  if (e < E0) { s = ei[e]; d = ei[E0 + e]; }
  else        { s = e - E0; d = e - E0; }    // self loop
}

// ---------------- fills / conversions ----------------
__global__ void k_fill(float* __restrict__ p, long long n, float v) {
  long long i = (long long)blockIdx.x * blockDim.x + threadIdx.x;
  if (i < n) p[i] = v;
}

__global__ void k_cvt_x(const float* __restrict__ x, unsigned short* __restrict__ xb) {
  int i = blockIdx.x * blockDim.x + threadIdx.x;          // NN*K1PAD
  if (i >= NN * K1PAD) return;
  int n = i / K1PAD, k = i % K1PAD;
  xb[i] = (k < C1N) ? f2bf(x[n * C1N + k]) : (unsigned short)0;
}

// W [K][N] fp32 row-major  ->  Wt [Npad][Kpad] bf16 (transposed + zero-padded)
__global__ void k_cvt_wt(const float* __restrict__ w, unsigned short* __restrict__ wt,
                         int K, int N, int Kpad, int Npad) {
  int i = blockIdx.x * blockDim.x + threadIdx.x;
  if (i >= Npad * Kpad) return;
  int n = i / Kpad, k = i % Kpad;
  wt[i] = (n < N && k < K) ? f2bf(w[k * N + n]) : (unsigned short)0;
}

// elu'd layer-1 features fp32 [NN][D1N] -> bf16 [NN][K2PAD]
__global__ void k_cvt_h2(const float* __restrict__ f, unsigned short* __restrict__ hb) {
  int i = blockIdx.x * blockDim.x + threadIdx.x;          // NN*K2PAD
  if (i >= NN * K2PAD) return;
  int n = i / K2PAD, k = i % K2PAD;
  hb[i] = (k < D1N) ? f2bf(f[n * D1N + k]) : (unsigned short)0;
}

// ---- bf16 WMMA GEMM: C[M][Npad] = A[M][Kpad] x Bt[Npad][Kpad]^T --------------------
// One wave computes a 16 x (16*NTW) slab: the A fragment for each K-step is loaded once
// and reused across NTW WMMAs (4x less A traffic, higher wmma:vmem ratio).
// Fragments follow the CDNA5 16-bit WMMA layouts:
//   A 16x32: lane l (m=l&15) holds K = (l>=16? 8:0)+[0..7] in elems 0..7 and +16 in 8..15
//   B 32x16: lane l (n=l&15) holds K = (l>=16? 8:0)+[0..7] in elems 0..7 and +16 in 8..15 (B pre-transposed)
__global__ void k_gemm_bf16(const unsigned short* __restrict__ A,
                            const unsigned short* __restrict__ Bt,
                            float* __restrict__ C,
                            int Mtiles, int Ngroups, int Kpad, int Npad) {
  int wave = (int)((blockIdx.x * blockDim.x + threadIdx.x) >> 5);
  int lane = threadIdx.x & 31;
  if (wave >= Mtiles * Ngroups) return;
  int tm  = (wave / Ngroups) * 16;
  int tn0 = (wave % Ngroups) * (16 * NTW);
  int half = lane >> 4;
  int l16  = lane & 15;

  const unsigned short* arow = A + (size_t)(tm + l16) * Kpad + half * 8;

  v8f acc[NTW];
#pragma unroll
  for (int t = 0; t < NTW; ++t) acc[t] = (v8f){};

  for (int k0 = 0; k0 < Kpad; k0 += 32) {
    v16bf a;
    ((uint4*)&a)[0] = *(const uint4*)(arow + k0);        // K = k0+half*8    .. +7
    ((uint4*)&a)[1] = *(const uint4*)(arow + k0 + 16);   // K = k0+16+half*8 .. +7
#pragma unroll
    for (int t = 0; t < NTW; ++t) {
      const unsigned short* brow =
          Bt + (size_t)(tn0 + t * 16 + l16) * Kpad + half * 8;
      v16bf b;
      ((uint4*)&b)[0] = *(const uint4*)(brow + k0);      // K = k0+half*8    .. +7
      ((uint4*)&b)[1] = *(const uint4*)(brow + k0 + 16); // K = k0+16+half*8 .. +7
      acc[t] = __builtin_amdgcn_wmma_f32_16x16x32_bf16(
          /*neg_a=*/false, a, /*neg_b=*/false, b,
          /*c_mod=*/(short)0, acc[t], /*reuse_a=*/false, /*reuse_b=*/false);
    }
  }
  // C/D layout: VGPR r -> row (r + 8*half), col = l16
#pragma unroll
  for (int t = 0; t < NTW; ++t) {
    float* crow = C + (size_t)(tm + half * 8) * Npad + tn0 + t * 16 + l16;
#pragma unroll
    for (int r = 0; r < 8; ++r) crow[(size_t)r * Npad] = acc[t][r];
  }
}

// ---------------- attention logits per node/head ----------------
__global__ void k_alpha(const float* __restrict__ Hm, int ldh, int heads, int C,
                        const float* __restrict__ asrc, const float* __restrict__ adst,
                        float* __restrict__ as_, float* __restrict__ ad_) {
  int i = blockIdx.x * blockDim.x + threadIdx.x;          // NN*heads
  if (i >= NN * heads) return;
  int n = i / heads, h = i % heads;
  const float* hp = Hm + (size_t)n * ldh + h * C;
  const float* sp = asrc + h * C;
  const float* dp = adst + h * C;
  float ss = 0.f, dd = 0.f;
  for (int c = 0; c < C; ++c) { float v = hp[c]; ss += v * sp[c]; dd += v * dp[c]; }
  as_[i] = ss; ad_[i] = dd;
}

// ---------------- edge passes (segment softmax + aggregation) ----------------
__global__ void k_edge_max(const int* __restrict__ ei, int heads,
                           const float* __restrict__ as_, const float* __restrict__ ad_,
                           float* __restrict__ m) {
  int i = blockIdx.x * blockDim.x + threadIdx.x;          // E_TOT*heads
  if (i >= E_TOT * heads) return;
  int e = i / heads, h = i % heads;
  int s, d; edge_sd(ei, e, s, d);
  float v = as_[s * heads + h] + ad_[d * heads + h];
  v = (v > 0.f) ? v : NEG_SLOPE * v;
  atomicMaxF(&m[d * heads + h], v);
}

__global__ void k_edge_exp(const int* __restrict__ ei, int heads,
                           const float* __restrict__ as_, const float* __restrict__ ad_,
                           const float* __restrict__ m,
                           float* __restrict__ ew, float* __restrict__ denom) {
  int i = blockIdx.x * blockDim.x + threadIdx.x;
  if (i >= E_TOT * heads) return;
  int e = i / heads, h = i % heads;
  int s, d; edge_sd(ei, e, s, d);
  float v = as_[s * heads + h] + ad_[d * heads + h];
  v = (v > 0.f) ? v : NEG_SLOPE * v;
  float ex = __expf(v - m[d * heads + h]);
  ew[i] = ex;
  atomicAdd(&denom[d * heads + h], ex);
}

// normalize coefficients in-place: ew <- ew / (denom[dst] + eps)
__global__ void k_edge_norm(const int* __restrict__ ei, int heads,
                            const float* __restrict__ denom, float* __restrict__ ew) {
  int i = blockIdx.x * blockDim.x + threadIdx.x;
  if (i >= E_TOT * heads) return;
  int e = i / heads, h = i % heads;
  int s, d; edge_sd(ei, e, s, d); (void)s;
  ew[i] = ew[i] / (denom[d * heads + h] + 1e-16f);
}

// out[dst] += coef * H[src], 4 channels per thread (float4 feature load)
__global__ void k_edge_aggr(const int* __restrict__ ei, int heads, int Cper, int Ctot,
                            const float* __restrict__ coef,
                            const float* __restrict__ Hm, int ldh,
                            float* __restrict__ out, int ldo) {
  int c4n = Ctot >> 2;                                    // Ctot % 4 == 0 (780, 100)
  long long i = (long long)blockIdx.x * blockDim.x + threadIdx.x;  // E_TOT * c4n
  if (i >= (long long)E_TOT * c4n) return;
  int e = (int)(i / c4n);
  int c = (int)(i % c4n) * 4;
  int s, d; edge_sd(ei, e, s, d);
  float4 hv = *(const float4*)(Hm + (size_t)s * ldh + c);
  const float hvv[4] = {hv.x, hv.y, hv.z, hv.w};
  int h = c / Cper;                                       // at most one head boundary in group
  float cf0 = coef[(size_t)e * heads + h];
  float* op = out + (size_t)d * ldo + c;
#pragma unroll
  for (int j = 0; j < 4; ++j) {
    bool nxt = (c + j) >= (h + 1) * Cper;
    float cf = nxt ? coef[(size_t)e * heads + h + 1] : cf0;
    atomicAdd(op + j, hvv[j] * cf);
  }
}

// ---------------- bias + activation (0 = ELU, 1 = ReLU) ----------------
__global__ void k_bias_act(float* __restrict__ out, const float* __restrict__ b, int C, int act) {
  long long i = (long long)blockIdx.x * blockDim.x + threadIdx.x;  // NN*C
  if (i >= (long long)NN * C) return;
  int c = (int)(i % C);
  float v = out[i] + b[c];
  out[i] = act ? fmaxf(v, 0.f) : (v > 0.f ? v : (__expf(v) - 1.f));
}

// ---------------- global max pool ----------------
__global__ void k_pool(const float* __restrict__ h, const int* __restrict__ batch,
                       float* __restrict__ pooled) {
  int i = blockIdx.x * blockDim.x + threadIdx.x;          // NN*C2N
  if (i >= NN * C2N) return;
  int n = i / C2N, c = i % C2N;
  atomicMaxF(&pooled[batch[n] * C2N + c], h[i]);          // h already >= 0 (relu)
}

// ---------------- final dense + relu ----------------
__global__ void k_final(const float* __restrict__ pooled, const float* __restrict__ Wg,
                        const float* __restrict__ bg, float* __restrict__ out) {
  int i = blockIdx.x * blockDim.x + threadIdx.x;          // NG*C2N
  if (i >= NG * C2N) return;
  int g = i / C2N, c = i % C2N;
  float acc = bg[c];
  for (int k = 0; k < C2N; ++k) acc += pooled[g * C2N + k] * Wg[k * C2N + c];
  out[i] = fmaxf(acc, 0.f);
}

// ---------------- host side ----------------
static inline int cdiv(long long a, long long b) { return (int)((a + b - 1) / b); }

extern "C" void kernel_launch(void* const* d_in, const int* in_sizes, int n_in,
                              void* d_out, int out_size, void* d_ws, size_t ws_size,
                              hipStream_t stream) {
  const float* x     = (const float*)d_in[0];
  const int*   ei    = (const int*)  d_in[1];
  const int*   batch = (const int*)  d_in[2];
  const float* W1    = (const float*)d_in[4];
  const float* as1w  = (const float*)d_in[5];
  const float* ad1w  = (const float*)d_in[6];
  const float* b1    = (const float*)d_in[7];
  const float* W2    = (const float*)d_in[8];
  const float* as2w  = (const float*)d_in[9];
  const float* ad2w  = (const float*)d_in[10];
  const float* b2    = (const float*)d_in[11];
  const float* Wg    = (const float*)d_in[12];
  const float* bg    = (const float*)d_in[13];
  float* out = (float*)d_out;

  // ---- workspace layout (bump allocator, 256B aligned, with overlays) ----
  char* ws = (char*)d_ws;
  size_t o = 0;
  auto alloc = [&](size_t bytes) { size_t r = o; o += (bytes + 255) & ~(size_t)255; return r; };
  size_t oXB  = alloc((size_t)NN * K1PAD * 2);       // bf16 x
  size_t oW1T = alloc((size_t)D1PAD * K1PAD * 2);    // bf16 W1^T  [832][96]
  size_t oW2T = alloc((size_t)N2PAD * K2PAD * 2);    // bf16 W2^T  [128][800]
  size_t oH1  = alloc((size_t)NN * D1PAD * 4);       // f32 h1 (66.6MB); later hosts HB2 + H2
  size_t oO1  = alloc((size_t)NN * D1N * 4);         // f32 aggregated/elu features
  size_t oAS1 = alloc((size_t)NN * H1N * 4);
  size_t oAD1 = alloc((size_t)NN * H1N * 4);
  size_t oM1  = alloc((size_t)NN * H1N * 4);
  size_t oDN1 = alloc((size_t)NN * H1N * 4);
  size_t oEW1 = alloc((size_t)E_TOT * H1N * 4);      // 13.6MB; later hosts layer-2 smalls

  unsigned short* XB  = (unsigned short*)(ws + oXB);
  unsigned short* W1T = (unsigned short*)(ws + oW1T);
  unsigned short* W2T = (unsigned short*)(ws + oW2T);
  float* H1  = (float*)(ws + oH1);
  float* O1  = (float*)(ws + oO1);
  float* AS1 = (float*)(ws + oAS1);
  float* AD1 = (float*)(ws + oAD1);
  float* M1  = (float*)(ws + oM1);
  float* DN1 = (float*)(ws + oDN1);
  float* EW1 = (float*)(ws + oEW1);
  // overlays: H1 region after layer-1 aggregation (H1 is dead once O1 is built)
  unsigned short* HB2 = (unsigned short*)(ws + oH1);                      // 32MB
  float* H2 = (float*)(ws + oH1 + (size_t)40 * 1024 * 1024);              // 10.24MB, disjoint from HB2
  // overlays: EW1 region after layer-1 aggregation
  float* AS2    = (float*)(ws + oEW1 + 0);
  float* AD2    = (float*)(ws + oEW1 + 81920);
  float* M2     = (float*)(ws + oEW1 + 163840);
  float* DN2    = (float*)(ws + oEW1 + 245760);
  float* EW2    = (float*)(ws + oEW1 + 327680);                           // 1.36MB
  float* O2     = (float*)(ws + oEW1 + 1703936);                          // 8MB
  float* POOLED = (float*)(ws + oEW1 + 9703936);                          // 0.2MB

  const int B = 256;

  // ---- stage: convert inputs to bf16 ----
  k_cvt_x <<<cdiv((long long)NN * K1PAD, B), B, 0, stream>>>(x, XB);
  k_cvt_wt<<<cdiv((long long)D1PAD * K1PAD, B), B, 0, stream>>>(W1, W1T, C1N, D1N, K1PAD, D1PAD);
  k_cvt_wt<<<cdiv((long long)N2PAD * K2PAD, B), B, 0, stream>>>(W2, W2T, D1N, C2N, K2PAD, N2PAD);

  // ---- layer 1 GEMM: H1 = x @ W1 ----
  {
    int mt = NN / 16, ngr = D1PAD / (16 * NTW);
    k_gemm_bf16<<<cdiv((long long)mt * ngr * 32, B), B, 0, stream>>>(XB, W1T, H1, mt, ngr, K1PAD, D1PAD);
  }
  // ---- layer 1 attention ----
  k_alpha<<<cdiv((long long)NN * H1N, B), B, 0, stream>>>(H1, D1PAD, H1N, C1N, as1w, ad1w, AS1, AD1);
  k_fill <<<cdiv((long long)NN * H1N, B), B, 0, stream>>>(M1,  (long long)NN * H1N, -3.0e38f);
  k_fill <<<cdiv((long long)NN * H1N, B), B, 0, stream>>>(DN1, (long long)NN * H1N, 0.f);
  k_fill <<<cdiv((long long)NN * D1N, B), B, 0, stream>>>(O1,  (long long)NN * D1N, 0.f);
  k_edge_max <<<cdiv((long long)E_TOT * H1N, B), B, 0, stream>>>(ei, H1N, AS1, AD1, M1);
  k_edge_exp <<<cdiv((long long)E_TOT * H1N, B), B, 0, stream>>>(ei, H1N, AS1, AD1, M1, EW1, DN1);
  k_edge_norm<<<cdiv((long long)E_TOT * H1N, B), B, 0, stream>>>(ei, H1N, DN1, EW1);
  k_edge_aggr<<<cdiv((long long)E_TOT * (D1N / 4), B), B, 0, stream>>>(ei, H1N, C1N, D1N, EW1, H1, D1PAD, O1, D1N);
  k_bias_act<<<cdiv((long long)NN * D1N, B), B, 0, stream>>>(O1, b1, D1N, /*elu*/0);

  // ---- layer 2 GEMM: H2 = elu(h1) @ W2 ----
  k_cvt_h2<<<cdiv((long long)NN * K2PAD, B), B, 0, stream>>>(O1, HB2);
  {
    int mt = NN / 16, ngr = N2PAD / (16 * NTW);
    k_gemm_bf16<<<cdiv((long long)mt * ngr * 32, B), B, 0, stream>>>(HB2, W2T, H2, mt, ngr, K2PAD, N2PAD);
  }
  // ---- layer 2 attention ----
  k_alpha<<<cdiv((long long)NN, B), B, 0, stream>>>(H2, N2PAD, 1, C2N, as2w, ad2w, AS2, AD2);
  k_fill <<<cdiv((long long)NN, B), B, 0, stream>>>(M2,  (long long)NN, -3.0e38f);
  k_fill <<<cdiv((long long)NN, B), B, 0, stream>>>(DN2, (long long)NN, 0.f);
  k_fill <<<cdiv((long long)NN * C2N, B), B, 0, stream>>>(O2, (long long)NN * C2N, 0.f);
  k_fill <<<cdiv((long long)NG * C2N, B), B, 0, stream>>>(POOLED, (long long)NG * C2N, 0.f);
  k_edge_max <<<cdiv((long long)E_TOT, B), B, 0, stream>>>(ei, 1, AS2, AD2, M2);
  k_edge_exp <<<cdiv((long long)E_TOT, B), B, 0, stream>>>(ei, 1, AS2, AD2, M2, EW2, DN2);
  k_edge_norm<<<cdiv((long long)E_TOT, B), B, 0, stream>>>(ei, 1, DN2, EW2);
  k_edge_aggr<<<cdiv((long long)E_TOT * (C2N / 4), B), B, 0, stream>>>(ei, 1, C2N, C2N, EW2, H2, N2PAD, O2, C2N);
  k_bias_act<<<cdiv((long long)NN * C2N, B), B, 0, stream>>>(O2, b2, C2N, /*relu*/1);

  // ---- pool + final dense ----
  k_pool <<<cdiv((long long)NN * C2N, B), B, 0, stream>>>(O2, batch, POOLED);
  k_final<<<cdiv((long long)NG * C2N, B), B, 0, stream>>>(POOLED, Wg, bg, out);

  (void)in_sizes; (void)n_in; (void)out_size; (void)ws_size;
}
